// HGAT_3530463117779
// MI455X (gfx1250) — matrix operand order repaired and checked by reading, not compiled, not run on hardware
//
#include <hip/hip_runtime.h>
#include <hip/hip_bf16.h>

// ---------------------------------------------------------------------------
// HGAT pipeline for MI455X (gfx1250, wave32, WMMA bf16)
// ---------------------------------------------------------------------------

typedef __attribute__((ext_vector_type(16))) __bf16 v16bf;
typedef __attribute__((ext_vector_type(8)))  float  v8f;

#define GH 4     // heads
#define GC 32    // channels/head
#define TB 256   // threads per block (8 waves)

// float -> bf16 (round to nearest even)
__device__ __forceinline__ unsigned short f2bf(float f) {
  unsigned int u = __float_as_uint(f);
  return (unsigned short)((u + 0x7FFFu + ((u >> 16) & 1u)) >> 16);
}

// orderable-float atomic max (works for +/-, init with -3e38)
__device__ __forceinline__ void atomicMaxF(float* addr, float val) {
  if (val >= 0.0f) atomicMax((int*)addr, __float_as_int(val));
  else             atomicMin((unsigned int*)addr, __float_as_uint(val));
}

// ---------------------------------------------------------------------------
__global__ __launch_bounds__(TB) void hgat_fill(float* __restrict__ p, float v, int n) {
  int i = blockIdx.x * TB + threadIdx.x;
  if (i < n) p[i] = v;
}

__global__ __launch_bounds__(TB) void hgat_gather(const float* __restrict__ emb,
                                                  const int* __restrict__ ids,
                                                  float* __restrict__ out, int N, int D) {
  int i = blockIdx.x * TB + threadIdx.x;
  if (i >= N * D) return;
  int n = i / D, d = i - n * D;
  out[i] = emb[(size_t)ids[n] * D + d];
}

// Pre-swizzle f32 weights [K][N] row-major into the WMMA B layout these GEMMs use (contiguous k, same order as A):
// out[((t*kTiles + kt)*32 + lane)*16 + i] = bf16( W[kt*32 + i + 16*(lane>>4)][t*16 + (lane&15)] )
__global__ __launch_bounds__(TB) void hgat_swz(const float* __restrict__ W,
                                               unsigned short* __restrict__ out,
                                               int K, int N) {
  int idx = blockIdx.x * TB + threadIdx.x;
  if (idx >= K * N) return;
  int i  = idx & 15;
  int l  = (idx >> 4) & 31;
  int g  = idx >> 9;                 // (t*kTiles + kt)
  int kTiles = K >> 5;
  int kt = g % kTiles;
  int t  = g / kTiles;
  int k  = kt * 32 + i + 16 * (l >> 4);
  int n  = t * 16 + (l & 15);
  out[idx] = f2bf(W[(size_t)k * N + n]);
}

// watt[h*K + k] = sum_c W[k][h*32+c] * a[h*32+c]   (folds dst/src projection into K-vec)
__global__ __launch_bounds__(TB) void hgat_watt(const float* __restrict__ W,
                                                const float* __restrict__ a,
                                                float* __restrict__ watt, int K, int HCc) {
  int i = blockIdx.x * TB + threadIdx.x;
  if (i >= GH * K) return;
  int h = i / K, k = i - h * K;
  float s = 0.f;
  #pragma unroll
  for (int c = 0; c < GC; ++c) s += W[(size_t)k * HCc + h * GC + c] * a[h * GC + c];
  watt[i] = s;   // layout [h][k]
}

// out[n*4+h] = X[n,:] . watt[h,:]
__global__ __launch_bounds__(TB) void hgat_ascore(const float* __restrict__ X,
                                                  const float* __restrict__ watt,
                                                  float* __restrict__ out, int N, int K) {
  int i = blockIdx.x * TB + threadIdx.x;
  if (i >= N * GH) return;
  int n = i >> 2, h = i & 3;
  const float4* xr = (const float4*)(X + (size_t)n * K);
  const float4* wr = (const float4*)(watt + (size_t)h * K);
  float s = 0.f;
  for (int k = 0; k < (K >> 2); ++k) {
    float4 xa = xr[k], wa = wr[k];
    s += xa.x * wa.x + xa.y * wa.y + xa.z * wa.z + xa.w * wa.w;
  }
  out[i] = s;
}

// ---------------------------------------------------------------------------
// Templated WMMA GEMM: one wave computes a full 16-row x N strip.
// A (f32->bf16) loaded/converted once per k-step, reused across all NT tiles.
// K = KT*32, N = NT*16, M % 16 == 0.
template <int KT, int NT>
__global__ __launch_bounds__(TB) void hgat_gemm_t(const float* __restrict__ X,
                                                  const unsigned short* __restrict__ Bsw,
                                                  float* __restrict__ Out,
                                                  const float* __restrict__ bias,
                                                  int M) {
  const int lane = threadIdx.x & 31;
  const int wave = blockIdx.x * (TB >> 5) + (threadIdx.x >> 5);
  if (wave >= (M >> 4)) return;
  const int laneLo = lane & 15, laneHi = lane >> 4;
  constexpr int K = KT * 32;
  constexpr int N = NT * 16;

  v8f acc[NT];
  #pragma unroll
  for (int t = 0; t < NT; ++t) acc[t] = (v8f){};

  const float* xr = X + (size_t)(wave * 16 + laneLo) * K;
  const unsigned short* bl = Bsw + (size_t)lane * 16;

  #pragma unroll
  for (int kt = 0; kt < KT; ++kt) {
    // A rows: lane holds M=laneLo; K = {16*laneHi .. +15}
    const float* pa = xr + kt * 32 + laneHi * 16;
    float4 a0 = *(const float4*)(pa);
    float4 a1 = *(const float4*)(pa + 4);
    float4 a2 = *(const float4*)(pa + 8);
    float4 a3 = *(const float4*)(pa + 12);
    union { v16bf v; unsigned short s[16]; } ua;
    ua.s[0]  = f2bf(a0.x); ua.s[1]  = f2bf(a0.y); ua.s[2]  = f2bf(a0.z); ua.s[3]  = f2bf(a0.w);
    ua.s[4]  = f2bf(a1.x); ua.s[5]  = f2bf(a1.y); ua.s[6]  = f2bf(a1.z); ua.s[7]  = f2bf(a1.w);
    ua.s[8]  = f2bf(a2.x); ua.s[9]  = f2bf(a2.y); ua.s[10] = f2bf(a2.z); ua.s[11] = f2bf(a2.w);
    ua.s[12] = f2bf(a3.x); ua.s[13] = f2bf(a3.y); ua.s[14] = f2bf(a3.z); ua.s[15] = f2bf(a3.w);

    #pragma unroll
    for (int t = 0; t < NT; ++t) {
      v16bf bv = *(const v16bf*)(bl + (size_t)(t * KT + kt) * 512);
      acc[t] = __builtin_amdgcn_wmma_f32_16x16x32_bf16(false, ua.v, false, bv,
                                                       (short)0, acc[t], false, false);
    }
  }

  #pragma unroll
  for (int t = 0; t < NT; ++t) {
    const int col = t * 16 + laneLo;
    const float bz = bias ? bias[col] : 0.f;
    float* orow = Out + (size_t)(wave * 16 + 8 * laneHi) * N + col;
    #pragma unroll
    for (int r = 0; r < 8; ++r) orow[(size_t)r * N] = acc[t][r] + bz;
  }
}

// Dynamic-shape fallback (one wave per 16x16 tile)
__global__ __launch_bounds__(TB) void hgat_gemm(const float* __restrict__ X,
                                                const unsigned short* __restrict__ Bsw,
                                                float* __restrict__ Out,
                                                const float* __restrict__ bias,
                                                int M, int K, int N) {
  const int lane   = threadIdx.x & 31;
  const int wave   = blockIdx.x * (TB >> 5) + (threadIdx.x >> 5);
  const int ntiles = N >> 4;
  const int mt = wave / ntiles;
  const int nt = wave - mt * ntiles;
  if (mt >= (M >> 4)) return;
  const int laneLo = lane & 15, laneHi = lane >> 4;
  const int kTiles = K >> 5;

  v8f acc = {};
  const float* xr = X + (size_t)(mt * 16 + laneLo) * K;
  const unsigned short* bp = Bsw + ((size_t)nt * kTiles) * 512;
  union { v16bf v; unsigned short s[16]; } ua;

  for (int kt = 0; kt < kTiles; ++kt) {
    const float* pa = xr + kt * 32 + laneHi * 16;
    float4 a0 = *(const float4*)(pa);
    float4 a1 = *(const float4*)(pa + 4);
    float4 a2 = *(const float4*)(pa + 8);
    float4 a3 = *(const float4*)(pa + 12);
    ua.s[0]  = f2bf(a0.x); ua.s[1]  = f2bf(a0.y); ua.s[2]  = f2bf(a0.z); ua.s[3]  = f2bf(a0.w);
    ua.s[4]  = f2bf(a1.x); ua.s[5]  = f2bf(a1.y); ua.s[6]  = f2bf(a1.z); ua.s[7]  = f2bf(a1.w);
    ua.s[8]  = f2bf(a2.x); ua.s[9]  = f2bf(a2.y); ua.s[10] = f2bf(a2.z); ua.s[11] = f2bf(a2.w);
    ua.s[12] = f2bf(a3.x); ua.s[13] = f2bf(a3.y); ua.s[14] = f2bf(a3.z); ua.s[15] = f2bf(a3.w);
    v16bf bv = *(const v16bf*)(bp + (size_t)kt * 512 + lane * 16);
    acc = __builtin_amdgcn_wmma_f32_16x16x32_bf16(false, ua.v, false, bv,
                                                  (short)0, acc, false, false);
  }

  const int col = nt * 16 + laneLo;
  const float bz = bias ? bias[col] : 0.f;
  float* orow = Out + (size_t)(mt * 16 + 8 * laneHi) * N + col;
  #pragma unroll
  for (int r = 0; r < 8; ++r) orow[(size_t)r * N] = acc[r] + bz;
}

// pass 1: e = leaky_relu(a_s[src]+a_d[dst], 0.2); segment max via atomics
__global__ __launch_bounds__(TB) void hgat_edge1(const int* __restrict__ src,
                                                 const int* __restrict__ dst,
                                                 const float* __restrict__ asc,
                                                 const float* __restrict__ adt,
                                                 float* __restrict__ ebuf,
                                                 float* __restrict__ mmax, int E) {
  int i = blockIdx.x * TB + threadIdx.x;
  if (i >= E * GH) return;
  int e = i >> 2, h = i & 3;
  int s = src[e], d = dst[e];
  float v = asc[s * 4 + h] + adt[d * 4 + h];
  v = v > 0.f ? v : 0.2f * v;
  ebuf[i] = v;
  atomicMaxF(&mmax[d * 4 + h], v);
}

// pass 2: ex = exp(e - m[dst]); denom[dst] += ex; acc[dst] += ex*hs[src]
__global__ __launch_bounds__(TB) void hgat_edge2(const int* __restrict__ src,
                                                 const int* __restrict__ dst,
                                                 const float* __restrict__ ebuf,
                                                 const float* __restrict__ mmax,
                                                 const float* __restrict__ hs,
                                                 float* __restrict__ denom,
                                                 float* __restrict__ acc, int E) {
  int i = blockIdx.x * TB + threadIdx.x;
  if (i >= E * 128) return;
  int e = i >> 7, r = i & 127, h = r >> 5;
  int s = src[e], d = dst[e];
  float ex = expf(ebuf[e * 4 + h] - mmax[d * 4 + h]);
  if ((r & 31) == 0) atomicAdd(&denom[d * 4 + h], ex);
  atomicAdd(&acc[(size_t)d * 128 + r], ex * hs[(size_t)s * 128 + r]);
}

// normalize + bias + leaky_relu(0.01); in-place safe
__global__ __launch_bounds__(TB) void hgat_fin(float* __restrict__ acc,
                                               const float* __restrict__ denom,
                                               const float* __restrict__ bias,
                                               float* __restrict__ out, int N) {
  int i = blockIdx.x * TB + threadIdx.x;
  if (i >= N * 128) return;
  int n = i >> 7, r = i & 127, h = r >> 5;
  float v = acc[i] / (denom[n * 4 + h] + 1e-16f) + bias[r];
  out[i] = v > 0.f ? v : 0.01f * v;
}

// ---------------------------------------------------------------------------
static inline int cdiv(int a, int b) { return (a + b - 1) / b; }

extern "C" void kernel_launch(void* const* d_in, const int* in_sizes, int n_in,
                              void* d_out, int out_size, void* d_ws, size_t ws_size,
                              hipStream_t stream) {
  (void)n_in; (void)out_size; (void)ws_size;
  const int*   user_ids  = (const int*)d_in[0];
  const int*   movie_ids = (const int*)d_in[1];
  const int*   um_src    = (const int*)d_in[2];
  const int*   um_dst    = (const int*)d_in[3];
  const int*   mu_src    = (const int*)d_in[4];
  const int*   mu_dst    = (const int*)d_in[5];
  const float* user_emb  = (const float*)d_in[6];
  const float* movie_emb = (const float*)d_in[7];
  const float* l1_um_Wsrc = (const float*)d_in[8];
  const float* l1_um_Wdst = (const float*)d_in[9];
  const float* l1_um_asrc = (const float*)d_in[10];
  const float* l1_um_adst = (const float*)d_in[11];
  const float* l1_um_b    = (const float*)d_in[12];
  const float* l1_mu_Wsrc = (const float*)d_in[13];
  const float* l1_mu_Wdst = (const float*)d_in[14];
  const float* l1_mu_asrc = (const float*)d_in[15];
  const float* l1_mu_adst = (const float*)d_in[16];
  const float* l1_mu_b    = (const float*)d_in[17];
  const float* l2_um_W    = (const float*)d_in[18];
  const float* l2_um_asrc = (const float*)d_in[19];
  const float* l2_um_adst = (const float*)d_in[20];
  const float* l2_um_b    = (const float*)d_in[21];
  const float* l2_mu_W    = (const float*)d_in[22];
  const float* l2_mu_asrc = (const float*)d_in[23];
  const float* l2_mu_adst = (const float*)d_in[24];
  const float* l2_mu_b    = (const float*)d_in[25];
  const float* Wo         = (const float*)d_in[26];
  const float* bo         = (const float*)d_in[27];

  const int Nu = in_sizes[0];
  const int Nm = in_sizes[1];
  const int E  = in_sizes[2];
  const int D  = in_sizes[6] / Nu;   // 64
  const int HC = in_sizes[12];       // 128
  const int O  = in_sizes[27];       // 64

  // ---- workspace layout (float units; every region is a multiple of 16) ----
  float* ws = (float*)d_ws;
  size_t off = 0;
  float* hsA   = ws + off; off += (size_t)Nu * HC;   // projected src features
  float* xu1   = ws + off; off += (size_t)Nu * HC;   // layer-1 user output
  float* xm1   = ws + off; off += (size_t)Nm * HC;   // layer-1 movie output
  float* ebuf  = ws + off; off += (size_t)E * GH;
  float* asc   = ws + off; off += (size_t)Nu * GH;
  float* adt   = ws + off; off += (size_t)Nu * GH;
  float* mmax  = ws + off; off += (size_t)Nu * GH;
  float* denom = ws + off; off += (size_t)Nu * GH;
  float* gxu   = ws + off; off += (size_t)Nu * D;
  float* gxm   = ws + off; off += (size_t)Nm * D;
  float* wattb = ws + off; off += (size_t)8 * GH * HC;  // 8 slots of [H][K<=HC]
  unsigned short* wswz = (unsigned short*)(ws + off);

  unsigned short* swz_l1umS = wswz;                       // D*HC
  unsigned short* swz_l1muS = swz_l1umS + (size_t)D * HC; // D*HC
  unsigned short* swz_l2um  = swz_l1muS + (size_t)D * HC; // HC*HC
  unsigned short* swz_l2mu  = swz_l2um  + (size_t)HC * HC;
  unsigned short* swz_wo    = swz_l2mu  + (size_t)HC * HC; // HC*O

  float* w_l1um_s = wattb + 0 * (size_t)GH * HC;
  float* w_l1um_d = wattb + 1 * (size_t)GH * HC;
  float* w_l1mu_s = wattb + 2 * (size_t)GH * HC;
  float* w_l1mu_d = wattb + 3 * (size_t)GH * HC;
  float* w_l2um_s = wattb + 4 * (size_t)GH * HC;
  float* w_l2um_d = wattb + 5 * (size_t)GH * HC;
  float* w_l2mu_s = wattb + 6 * (size_t)GH * HC;
  float* w_l2mu_d = wattb + 7 * (size_t)GH * HC;

  // ---- d_out layout: [out_user | out_movie | xu | xm] ----
  float* out_user  = (float*)d_out;
  float* out_movie = out_user  + (size_t)Nu * O;
  float* xu_out    = out_movie + (size_t)Nm * O;
  float* xm_out    = xu_out    + (size_t)Nu * HC;

  // templated-GEMM dispatcher
  auto gemm = [&](const float* X, const unsigned short* Bsw, float* Out,
                  const float* bias, int M, int K, int N) {
    int blocks = cdiv((M >> 4) * 32, TB);
    if (K == 64 && N == 128)
      hgat_gemm_t<2, 8><<<blocks, TB, 0, stream>>>(X, Bsw, Out, bias, M);
    else if (K == 128 && N == 128)
      hgat_gemm_t<4, 8><<<blocks, TB, 0, stream>>>(X, Bsw, Out, bias, M);
    else if (K == 128 && N == 64)
      hgat_gemm_t<4, 4><<<blocks, TB, 0, stream>>>(X, Bsw, Out, bias, M);
    else {
      int waves = (M >> 4) * (N >> 4);
      hgat_gemm<<<cdiv(waves * 32, TB), TB, 0, stream>>>(X, Bsw, Out, bias, M, K, N);
    }
  };

  // ---- prep: gather embeddings, swizzle weights, attention K-vectors ----
  hgat_gather<<<cdiv(Nu * D, TB), TB, 0, stream>>>(user_emb,  user_ids,  gxu, Nu, D);
  hgat_gather<<<cdiv(Nm * D, TB), TB, 0, stream>>>(movie_emb, movie_ids, gxm, Nm, D);

  hgat_swz<<<cdiv(D * HC, TB),  TB, 0, stream>>>(l1_um_Wsrc, swz_l1umS, D,  HC);
  hgat_swz<<<cdiv(D * HC, TB),  TB, 0, stream>>>(l1_mu_Wsrc, swz_l1muS, D,  HC);
  hgat_swz<<<cdiv(HC * HC, TB), TB, 0, stream>>>(l2_um_W,    swz_l2um,  HC, HC);
  hgat_swz<<<cdiv(HC * HC, TB), TB, 0, stream>>>(l2_mu_W,    swz_l2mu,  HC, HC);
  hgat_swz<<<cdiv(HC * O, TB),  TB, 0, stream>>>(Wo,         swz_wo,    HC, O);

  hgat_watt<<<cdiv(GH * D, TB),  TB, 0, stream>>>(l1_um_Wsrc, l1_um_asrc, w_l1um_s, D,  HC);
  hgat_watt<<<cdiv(GH * D, TB),  TB, 0, stream>>>(l1_um_Wdst, l1_um_adst, w_l1um_d, D,  HC);
  hgat_watt<<<cdiv(GH * D, TB),  TB, 0, stream>>>(l1_mu_Wsrc, l1_mu_asrc, w_l1mu_s, D,  HC);
  hgat_watt<<<cdiv(GH * D, TB),  TB, 0, stream>>>(l1_mu_Wdst, l1_mu_adst, w_l1mu_d, D,  HC);
  hgat_watt<<<cdiv(GH * HC, TB), TB, 0, stream>>>(l2_um_W,    l2_um_asrc, w_l2um_s, HC, HC);
  hgat_watt<<<cdiv(GH * HC, TB), TB, 0, stream>>>(l2_um_W,    l2_um_adst, w_l2um_d, HC, HC);
  hgat_watt<<<cdiv(GH * HC, TB), TB, 0, stream>>>(l2_mu_W,    l2_mu_asrc, w_l2mu_s, HC, HC);
  hgat_watt<<<cdiv(GH * HC, TB), TB, 0, stream>>>(l2_mu_W,    l2_mu_adst, w_l2mu_d, HC, HC);

  // ---- one bipartite GATConv ----
  auto conv = [&](const float* xsrc, int Ns, int K, const unsigned short* wS,
                  const float* wattS, const float* wattD, const float* xdst, int Nd,
                  const int* sIdx, const int* dIdx, const float* bias, float* outb) {
    gemm(xsrc, wS, hsA, nullptr, Ns, K, HC);
    hgat_ascore<<<cdiv(Ns * GH, TB), TB, 0, stream>>>(xsrc, wattS, asc, Ns, K);
    hgat_ascore<<<cdiv(Nd * GH, TB), TB, 0, stream>>>(xdst, wattD, adt, Nd, K);
    hgat_fill<<<cdiv(Nd * GH, TB), TB, 0, stream>>>(mmax, -3.0e38f, Nd * GH);
    hgat_fill<<<cdiv(Nd * GH, TB), TB, 0, stream>>>(denom, 0.0f, Nd * GH);
    hgat_fill<<<cdiv(Nd * HC, TB), TB, 0, stream>>>(outb, 0.0f, Nd * HC);
    hgat_edge1<<<cdiv(E * GH, TB), TB, 0, stream>>>(sIdx, dIdx, asc, adt, ebuf, mmax, E);
    hgat_edge2<<<cdiv(E * 128, TB), TB, 0, stream>>>(sIdx, dIdx, ebuf, mmax, hsA, denom, outb, E);
    hgat_fin<<<cdiv(Nd * HC, TB), TB, 0, stream>>>(outb, denom, bias, outb, Nd);
  };

  // layer 1
  conv(gxu, Nu, D, swz_l1umS, w_l1um_s, w_l1um_d, gxm, Nm, um_src, um_dst, l1_um_b, xm1);
  conv(gxm, Nm, D, swz_l1muS, w_l1mu_s, w_l1mu_d, gxu, Nu, mu_src, mu_dst, l1_mu_b, xu1);

  // layer 2 (shared W per conv; outputs go straight into d_out xu/xm regions)
  conv(xu1, Nu, HC, swz_l2um, w_l2um_s, w_l2um_d, xm1, Nm, um_src, um_dst, l2_um_b, xm_out);
  conv(xm1, Nm, HC, swz_l2mu, w_l2mu_s, w_l2mu_d, xu1, Nu, mu_src, mu_dst, l2_mu_b, xu_out);

  // output projection with bias
  gemm(xu_out, swz_wo, out_user,  bo, Nu, HC, O);
  gemm(xm_out, swz_wo, out_movie, bo, Nm, HC, O);
}
